// RotaryMultiheadAttention_27384711480197
// MI455X (gfx1250) — matrix operand order repaired and checked by reading, not compiled, not run on hardware
//
#include <hip/hip_runtime.h>
#include <hip/hip_bf16.h>

typedef __bf16 bf16;
typedef __attribute__((ext_vector_type(16))) __bf16 v16bf;
typedef __attribute__((ext_vector_type(8)))  __bf16 v8bf;
typedef __attribute__((ext_vector_type(8)))  float  v8f;
typedef __attribute__((ext_vector_type(4)))  float  v4f;

// ---------------------------------------------------------------------------
// CDNA5 async global->LDS copy (ASYNCcnt-tracked), GVS addressing:
//   saddr = uniform 64-bit base (SGPR pair), vaddr = per-lane byte offset,
//   %0    = per-lane LDS byte address (flat addr truncated to 32 bits).
// ---------------------------------------------------------------------------
__device__ __forceinline__ void async_copy_b128(void* lds_ptr,
                                                const void* gbase,
                                                unsigned gbyte_off) {
  unsigned lds_addr = (unsigned)(size_t)lds_ptr;  // LDS aperture: low 32 bits
  asm volatile("global_load_async_to_lds_b128 %0, %1, %2 offset:0"
               :
               : "v"(lds_addr), "v"(gbyte_off), "s"(gbase)
               : "memory");
}
__device__ __forceinline__ void wait_async0() {
  asm volatile("s_wait_asynccnt 0" ::: "memory");
}

// ---------------------------------------------------------------------------
// WMMA fragment helpers (CDNA5 wave32 layouts, 16x16x32 bf16)
// ---------------------------------------------------------------------------

// A matrix 16x32 (MxK), LDS row-major [M][K], ld elements (multiple of 8).
__device__ __forceinline__ v16bf frag_a_lds(const bf16* base, int ld) {
  const int lane = threadIdx.x & 31;
  const int row  = lane & 15;
  const int koff = (lane >> 4) << 3;  // 0 or 8
  const bf16* r = base + row * ld + koff;
  v8bf lo = *(const v8bf*)(r);
  v8bf hi = *(const v8bf*)(r + 16);
  v16bf f;
#pragma unroll
  for (int i = 0; i < 8; ++i) { f[i] = lo[i]; f[i + 8] = hi[i]; }
  return f;
}

// B matrix 32x16 (KxN), LDS stored TRANSPOSED row-major [N][K], ld elements.
__device__ __forceinline__ v16bf frag_b_lds(const bf16* base, int ld) {
  const int lane = threadIdx.x & 31;
  const int n    = lane & 15;
  const int koff = (lane >> 4) << 3;  // 0 or 8
  const bf16* r = base + n * ld + koff;
  v8bf lo = *(const v8bf*)(r);
  v8bf hi = *(const v8bf*)(r + 16);
  v16bf f;
#pragma unroll
  for (int i = 0; i < 8; ++i) { f[i] = lo[i]; f[i + 8] = hi[i]; }
  return f;
}

__device__ __forceinline__ v8f wmma_bf16(v16bf a, v16bf b, v8f c) {
  return __builtin_amdgcn_wmma_f32_16x16x32_bf16(
      false, a, false, b, (short)0, c, false, false);
}

// ---------------------------------------------------------------------------
// Tiled GEMM + bias: C[M][N] = A[M][K] @ B[K][N] + bias[N]   (f32 out)
// Block tile 128x128, BK=64, 256 threads = 8 waves in 4(M) x 2(N).
// ---------------------------------------------------------------------------
template <bool A_BF16>
__global__ __launch_bounds__(256) void gemm_bias_kernel(
    const void* __restrict__ Aptr, const float* __restrict__ B,
    const float* __restrict__ bias, float* __restrict__ C,
    int M, int N, int K) {
  constexpr int BM = 128, BN = 128, BK = 64;
  constexpr int LDA = BK + 8;  // 72 elements -> 144B rows (16B aligned frags)
  constexpr int LDB = BK + 8;
  __shared__ bf16 As[BM * LDA];
  __shared__ bf16 Bs[BN * LDB];  // transposed: [N][K]

  const int m0 = blockIdx.y * BM;
  const int n0 = blockIdx.x * BN;
  const int tid = threadIdx.x;
  const int wave = tid >> 5;
  const int wm = (wave >> 1) * 32;  // 0,32,64,96
  const int wn = (wave & 1) * 64;   // 0,64

  const v8f vzero = {0.f, 0.f, 0.f, 0.f, 0.f, 0.f, 0.f, 0.f};
  v8f acc[2][4];
#pragma unroll
  for (int i = 0; i < 2; ++i)
#pragma unroll
    for (int j = 0; j < 4; ++j) acc[i][j] = vzero;

  for (int k0 = 0; k0 < K; k0 += BK) {
    __syncthreads();
    // ---- stage A (128 x 64) as bf16 ----
    if constexpr (A_BF16) {
      const char* A = (const char*)Aptr;
#pragma unroll
      for (int i = tid; i < BM * BK / 8; i += 256) {  // 1024 chunks of 16B
        int r = i >> 3;
        int c8 = (i & 7) * 8;
        async_copy_b128(&As[r * LDA + c8], A,
                        (unsigned)(((m0 + r) * K + k0 + c8) * 2));
      }
    } else {
      const float* A = (const float*)Aptr;
#pragma unroll
      for (int i = tid; i < BM * BK / 4; i += 256) {  // 2048 float4
        int r = i >> 4;
        int c4 = (i & 15) * 4;
        v4f vv = *(const v4f*)(A + (size_t)(m0 + r) * K + k0 + c4);
        bf16* dst = &As[r * LDA + c4];
        dst[0] = (bf16)vv[0]; dst[1] = (bf16)vv[1];
        dst[2] = (bf16)vv[2]; dst[3] = (bf16)vv[3];
      }
    }
    // ---- stage B (64 x 128) transposed -> Bs[n][k] ----
#pragma unroll
    for (int i = tid; i < BK * BN / 4; i += 256) {  // 2048 float4
      int kk = i >> 5;
      int c4 = (i & 31) * 4;
      v4f vv = *(const v4f*)(B + (size_t)(k0 + kk) * N + n0 + c4);
#pragma unroll
      for (int j = 0; j < 4; ++j) Bs[(c4 + j) * LDB + kk] = (bf16)vv[j];
    }
    if constexpr (A_BF16) wait_async0();
    __syncthreads();

    // ---- compute 16 WMMAs per wave ----
#pragma unroll
    for (int kk = 0; kk < 2; ++kk) {
      v16bf a0 = frag_a_lds(&As[(wm + 0) * LDA + kk * 32], LDA);
      v16bf a1 = frag_a_lds(&As[(wm + 16) * LDA + kk * 32], LDA);
#pragma unroll
      for (int j = 0; j < 4; ++j) {
        v16bf b = frag_b_lds(&Bs[(wn + j * 16) * LDB + kk * 32], LDB);
        acc[0][j] = wmma_bf16(a0, b, acc[0][j]);
        acc[1][j] = wmma_bf16(a1, b, acc[1][j]);
      }
    }
  }

  // ---- epilogue: bias + store f32 (C layout: lane=col, vgpr v=row) ----
  const int lane = tid & 31;
  const int nlo = lane & 15;
  const int mhi = (lane >> 4) * 8;
#pragma unroll
  for (int i = 0; i < 2; ++i)
#pragma unroll
    for (int j = 0; j < 4; ++j) {
      int col = n0 + wn + j * 16 + nlo;
      float bv = bias[col];
#pragma unroll
      for (int v = 0; v < 8; ++v) {
        int row = m0 + wm + i * 16 + mhi + v;
        C[(size_t)row * N + col] = acc[i][j][v] + bv;
      }
    }
}

// ---------------------------------------------------------------------------
// RoPE + head split: qkv f32 [B*S][3*1024] -> Qh/Kh/Vh bf16 [B][H][S][64]
// ---------------------------------------------------------------------------
__global__ __launch_bounds__(256) void rope_split_kernel(
    const float* __restrict__ qkv, bf16* __restrict__ Qh,
    bf16* __restrict__ Kh, bf16* __restrict__ Vh) {
  const int idx = blockIdx.x * 256 + threadIdx.x;  // 2^21 total
  const int d = idx & 31;
  const int s = (idx >> 5) & 2047;
  const int h = (idx >> 16) & 15;
  const int b = idx >> 20;

  const size_t row = (size_t)b * 2048 + s;
  const float* base = qkv + row * 3072 + h * 64;
  float q1 = base[d], q2 = base[d + 32];
  float k1 = base[1024 + d], k2 = base[1024 + d + 32];
  float v1 = base[2048 + d], v2 = base[2048 + d + 32];

  // inv_freq = 10000^(-d/32) = exp(-d * ln(10000)/32)
  float freq = (float)s * __expf(-(float)d * 0.28782313662425f);
  float sn, cs;
  __sincosf(freq, &sn, &cs);

  const size_t o = (((size_t)(b * 16 + h) * 2048) + s) * 64 + d;
  Qh[o]      = (bf16)(q1 * cs - q2 * sn);
  Qh[o + 32] = (bf16)(q2 * cs + q1 * sn);
  Kh[o]      = (bf16)(k1 * cs - k2 * sn);
  Kh[o + 32] = (bf16)(k2 * cs + k1 * sn);
  Vh[o]      = (bf16)v1;
  Vh[o + 32] = (bf16)v2;
}

// ---------------------------------------------------------------------------
// Flash attention: per (b,h), 128 query rows per block, stream 64-row K/V
// blocks through LDS with online softmax. Out bf16 [B][S][H*64].
// ---------------------------------------------------------------------------
__global__ __launch_bounds__(256) void flash_attn_kernel(
    const bf16* __restrict__ Qh, const bf16* __restrict__ Kh,
    const bf16* __restrict__ Vh, bf16* __restrict__ Out) {
  constexpr int BQ = 128, BKV = 64, D = 64, S = 2048;
  constexpr int LQ = D + 8;  // 72 elements -> 144B rows (16B aligned frags)
  __shared__ bf16 Qs[BQ * LQ];
  __shared__ bf16 Ks[BKV * LQ];  // [t][d]
  __shared__ bf16 Vt[D * LQ];    // transposed: [d][t]
  __shared__ bf16 Ps[BQ * LQ];   // probabilities, [row][t]

  const int tid = threadIdx.x;
  const int lane = tid & 31;
  const int wave = tid >> 5;
  const int bh = blockIdx.y;  // b*16 + h
  const int q0 = blockIdx.x * BQ;

  const bf16* Qbase = Qh + (size_t)bh * S * D;
  const bf16* Kbase = Kh + (size_t)bh * S * D;
  const bf16* Vbase = Vh + (size_t)bh * S * D;

  // stage Q block (128x64) via async copies
#pragma unroll
  for (int i = tid; i < BQ * D / 8; i += 256) {
    int r = i >> 3;
    int c = (i & 7) * 8;
    async_copy_b128(&Qs[r * LQ + c], Qbase,
                    (unsigned)(((q0 + r) * D + c) * 2));
  }

  const int mrow = wave * 16;
  const int hl = lane >> 4;
  const int nl = lane & 15;
  const float scale = 0.125f;  // 1/sqrt(64)

  const v8f vzero = {0.f, 0.f, 0.f, 0.f, 0.f, 0.f, 0.f, 0.f};
  v8f o[4];
#pragma unroll
  for (int j = 0; j < 4; ++j) o[j] = vzero;
  float rm[8], rl[8];
#pragma unroll
  for (int v = 0; v < 8; ++v) { rm[v] = -1e30f; rl[v] = 0.f; }

  wait_async0();

  for (int j0 = 0; j0 < S; j0 += BKV) {
    // prefetch next K/V block (global_prefetch_b8), one line per thread
    if (j0 + BKV < S) {
      __builtin_prefetch(Kbase + (size_t)(j0 + BKV) * D + tid * 16, 0, 1);
      __builtin_prefetch(Vbase + (size_t)(j0 + BKV) * D + tid * 16, 0, 1);
    }
    __syncthreads();
    // stage K block [t][d] via async copies
#pragma unroll
    for (int i = tid; i < BKV * D / 8; i += 256) {
      int r = i >> 3;
      int c = (i & 7) * 8;
      async_copy_b128(&Ks[r * LQ + c], Kbase,
                      (unsigned)(((j0 + r) * D + c) * 2));
    }
    // stage V block transposed: Vt[d][t] (needs VGPR round-trip)
#pragma unroll
    for (int i = tid; i < BKV * D / 8; i += 256) {
      int t = i >> 3;
      int c = (i & 7) * 8;
      v8bf vv = *(const v8bf*)(Vbase + (size_t)(j0 + t) * D + c);
#pragma unroll
      for (int e = 0; e < 8; ++e) Vt[(c + e) * LQ + t] = vv[e];
    }
    wait_async0();
    __syncthreads();

    // ---- scores: S = Q(16x64) @ K^T(64x64) per wave ----
    v8f sacc[4];
#pragma unroll
    for (int j = 0; j < 4; ++j) sacc[j] = vzero;
#pragma unroll
    for (int kk = 0; kk < 2; ++kk) {
      v16bf a = frag_a_lds(&Qs[mrow * LQ + kk * 32], LQ);
#pragma unroll
      for (int j = 0; j < 4; ++j) {
        v16bf b = frag_b_lds(&Ks[(j * 16) * LQ + kk * 32], LQ);
        sacc[j] = wmma_bf16(a, b, sacc[j]);
      }
    }

    // ---- online softmax (row = vgpr v + 8*hl, col = lane&15) ----
    float pls[4][8];
#pragma unroll
    for (int v = 0; v < 8; ++v) {
      float mx = fmaxf(fmaxf(sacc[0][v], sacc[1][v]),
                       fmaxf(sacc[2][v], sacc[3][v])) * scale;
#pragma unroll
      for (int off = 1; off < 16; off <<= 1)
        mx = fmaxf(mx, __shfl_xor(mx, off, 32));
      float nm = fmaxf(rm[v], mx);
      float corr = __expf(rm[v] - nm);
      float ls = 0.f;
#pragma unroll
      for (int j = 0; j < 4; ++j) {
        float p = __expf(sacc[j][v] * scale - nm);
        pls[j][v] = p;
        ls += p;
      }
#pragma unroll
      for (int off = 1; off < 16; off <<= 1) ls += __shfl_xor(ls, off, 32);
      rl[v] = rl[v] * corr + ls;
      rm[v] = nm;
#pragma unroll
      for (int j = 0; j < 4; ++j) o[j][v] *= corr;
    }

    // ---- write P (bf16) to LDS in [row][t] for A-fragment reload ----
#pragma unroll
    for (int j = 0; j < 4; ++j)
#pragma unroll
      for (int v = 0; v < 8; ++v)
        Ps[(mrow + hl * 8 + v) * LQ + j * 16 + nl] = (bf16)pls[j][v];

    // ---- O += P(16x64) @ V(64x64) ----
#pragma unroll
    for (int kk = 0; kk < 2; ++kk) {
      v16bf a = frag_a_lds(&Ps[mrow * LQ + kk * 32], LQ);
#pragma unroll
      for (int j = 0; j < 4; ++j) {
        v16bf b = frag_b_lds(&Vt[(j * 16) * LQ + kk * 32], LQ);
        o[j] = wmma_bf16(a, b, o[j]);
      }
    }
  }

  // ---- normalize + store: Out[b][s][h*64 + d] (bf16) ----
  const int b = bh >> 4, h = bh & 15;
#pragma unroll
  for (int v = 0; v < 8; ++v) {
    float inv = rl[v] > 0.f ? 1.f / rl[v] : 0.f;
    int srow = q0 + mrow + hl * 8 + v;
    size_t orow = ((size_t)b * 2048 + srow) * 1024 + h * 64;
#pragma unroll
    for (int j = 0; j < 4; ++j)
      Out[orow + j * 16 + nl] = (bf16)(o[j][v] * inv);
  }
}

// ---------------------------------------------------------------------------
// Host launcher
// ---------------------------------------------------------------------------
extern "C" void kernel_launch(void* const* d_in, const int* in_sizes, int n_in,
                              void* d_out, int out_size, void* d_ws,
                              size_t ws_size, hipStream_t stream) {
  const float* x    = (const float*)d_in[0];  // [2,2048,1024]
  const float* Wqkv = (const float*)d_in[1];  // [1024,3072]
  const float* bqkv = (const float*)d_in[2];  // [3072]
  const float* Wout = (const float*)d_in[3];  // [1024,1024]
  const float* bout = (const float*)d_in[4];  // [1024]
  float* out = (float*)d_out;                 // [2,2048,1024] f32

  const int M = 4096;  // B*S
  const int E = 1024;
  const size_t headElems = (size_t)2 * 16 * 2048 * 64;

  char* ws = (char*)d_ws;
  float* qkv  = (float*)ws;  // 48 MB
  bf16* Qh    = (bf16*)(ws + (size_t)M * 3072 * 4);
  bf16* Kh    = Qh + headElems;
  bf16* Vh    = Kh + headElems;
  bf16* AttnO = Vh + headElems;  // [4096][1024]

  gemm_bias_kernel<false><<<dim3(3072 / 128, M / 128), 256, 0, stream>>>(
      x, Wqkv, bqkv, qkv, M, 3072, E);

  rope_split_kernel<<<(2u * 16 * 2048 * 32) / 256, 256, 0, stream>>>(
      qkv, Qh, Kh, Vh);

  flash_attn_kernel<<<dim3(2048 / 128, 32), 256, 0, stream>>>(
      Qh, Kh, Vh, AttnO);

  gemm_bias_kernel<true><<<dim3(E / 128, M / 128), 256, 0, stream>>>(
      AttnO, Wout, bout, out, M, E, E);
}
